// Model_75840532513296
// MI455X (gfx1250) — matrix operand order repaired and checked by reading, not compiled, not run on hardware
//
#include <hip/hip_runtime.h>

// MI455X / gfx1250, wave32. Pipeline:
//   prep:  Whh1,Whh2 -> pack4 f16 ; T -> rowmax Kr + exp-domain pack4 ETp (f16)
//   GEMM1: emb[nums] @ Wxh1 + b1  -> XW1   (WMMA f16->f32, 4 wmma/wave/K-chunk)
//   scan1: h_t = tanh(XW1[t] + h @ Whh1)   -> G   (1 WGP, 1024 thr split-K)
//   GEMM2: G @ Wxh2 + b2          -> XW2
//   scan2: -> H
//   GEMM3: H @ Wl + bl            -> O
//   CRF scan in exp domain: c' = o + Kr + M + log(ET @ exp(c-M)) -> out[0]=c[BOS]

typedef __attribute__((ext_vector_type(16))) _Float16 v16h;
typedef __attribute__((ext_vector_type(8)))  float    v8f;

#define DD   512
#define NSEQ 2048
#define VV   512
#define BOS_ 0
#define EOS_ 1

union F16Frag { v16h v; uint4 q[2]; };
union H4 { uint2 u; _Float16 h[4]; };

// ---------------------------------------------------------------- prep kernels
// dst[((i/4)*cols + j)*4 + i%4] = (f16)src[i*cols + j]   (pack 4 rows per column)
__global__ void k_pack4_f16(const float* __restrict__ src,
                            _Float16* __restrict__ dst, int rows, int cols) {
  int idx = blockIdx.x * blockDim.x + threadIdx.x;
  if (idx >= rows * cols) return;
  int i = idx / cols, j = idx % cols;
  dst[((((i >> 2) * cols) + j) << 2) + (i & 3)] = (_Float16)src[idx];
}

// Kr[i] = max_j T[i,j];  ETp packed over sum index j:
//   ETp[((j/4)*V + i)*4 + j%4] = exp(T[i,j] - Kr[i])
__global__ void k_crf_prep(const float* __restrict__ T,
                           _Float16* __restrict__ ETp,
                           float* __restrict__ Kr) {
  int i = blockIdx.x * blockDim.x + threadIdx.x;
  if (i >= VV) return;
  float m = -INFINITY;
  for (int j = 0; j < VV; ++j) m = fmaxf(m, T[i * VV + j]);
  Kr[i] = m;
  for (int j = 0; j < VV; ++j)
    ETp[((((j >> 2) * VV) + i) << 2) + (j & 3)] =
        (_Float16)__expf(T[i * VV + j] - m);
}

// ---------------------------------------------------------------- WMMA GEMM
// C[m,n] = sum_k Arow(m)[k] * W[k,n] + bias[n]
// Block tile 64(M) x 128(N), 8 waves, each wave owns 32x32 => 4 wmma / K-chunk.
#define BM 64
#define BN 128
#define BK 32
#define LDH 48  // LDS row stride in halves: 96B, multiple of 16B

__global__ void __launch_bounds__(256)
k_gemm_wmma(const float* __restrict__ A, const int* __restrict__ gatherIdx,
            const float* __restrict__ W, const float* __restrict__ bias,
            float* __restrict__ C, int M, int N, int K) {
  __shared__ __align__(16) _Float16 sA[BM][LDH];   // A tile, row-major [m][k]
  __shared__ __align__(16) _Float16 sBt[BN][LDH];  // B tile, transposed [n][k]

  const int tid  = threadIdx.x;   // 256 threads = 8 waves
  const int lane = tid & 31;
  const int wave = tid >> 5;
  const int wm   = wave >> 2;     // 0..1: 32-row band
  const int wn   = wave & 3;      // 0..3: 32-col band
  const int m0   = blockIdx.y * BM;
  const int n0   = blockIdx.x * BN;

  const int hi   = (lane < 16) ? 0 : 1;
  const int lmod = lane & 15;
  const int kbA  = hi ? 16 : 0;   // A lane<16 holds K0-15 at M=lane
  const int kbB  = hi ? 16 : 0;   // B lane<16 holds K0-15 at N=lane

  v8f acc00 = {}, acc01 = {}, acc10 = {}, acc11 = {};

  for (int k0 = 0; k0 < K; k0 += BK) {
    __syncthreads();
    // stage A (optional embedding gather), f32 -> f16; 64x32
    for (int e = tid; e < BM * BK; e += 256) {
      int r = e >> 5, c = e & 31;
      long row = gatherIdx ? (long)gatherIdx[m0 + r] : (long)(m0 + r);
      sA[r][c] = (_Float16)A[row * (long)K + (k0 + c)];
    }
    // stage B transposed: coalesced read of W[k][n], write sBt[n][k]; 32x128
    for (int e = tid; e < BK * BN; e += 256) {
      int kk = e >> 7, nn = e & 127;
      sBt[nn][kk] = (_Float16)W[(long)(k0 + kk) * N + (n0 + nn)];
    }
    __syncthreads();

    F16Frag a0, a1, b0, b1;
    const int rA0 = wm * 32 + lmod, rA1 = rA0 + 16;
    const int rB0 = wn * 32 + lmod, rB1 = rB0 + 16;
    a0.q[0] = *reinterpret_cast<const uint4*>(&sA[rA0][kbA]);
    a0.q[1] = *reinterpret_cast<const uint4*>(&sA[rA0][kbA + 8]);
    a1.q[0] = *reinterpret_cast<const uint4*>(&sA[rA1][kbA]);
    a1.q[1] = *reinterpret_cast<const uint4*>(&sA[rA1][kbA + 8]);
    b0.q[0] = *reinterpret_cast<const uint4*>(&sBt[rB0][kbB]);
    b0.q[1] = *reinterpret_cast<const uint4*>(&sBt[rB0][kbB + 8]);
    b1.q[0] = *reinterpret_cast<const uint4*>(&sBt[rB1][kbB]);
    b1.q[1] = *reinterpret_cast<const uint4*>(&sBt[rB1][kbB + 8]);

    acc00 = __builtin_amdgcn_wmma_f32_16x16x32_f16(false, a0.v, false, b0.v,
                                                   (short)0, acc00, false, false);
    acc01 = __builtin_amdgcn_wmma_f32_16x16x32_f16(false, a0.v, false, b1.v,
                                                   (short)0, acc01, false, false);
    acc10 = __builtin_amdgcn_wmma_f32_16x16x32_f16(false, a1.v, false, b0.v,
                                                   (short)0, acc10, false, false);
    acc11 = __builtin_amdgcn_wmma_f32_16x16x32_f16(false, a1.v, false, b1.v,
                                                   (short)0, acc11, false, false);
  }

  // C/D layout: VGPR v -> M = v + (lane<16 ? 0 : 8), N = lane%16
  const int n0w = n0 + wn * 32;
  const int m0w = m0 + wm * 32 + (hi ? 8 : 0);
  const float bv0 = bias ? bias[n0w + lmod]      : 0.0f;
  const float bv1 = bias ? bias[n0w + 16 + lmod] : 0.0f;
#pragma unroll
  for (int v = 0; v < 8; ++v) {
    C[(long)(m0w + v)      * N + (n0w + lmod)]      = acc00[v] + bv0;
    C[(long)(m0w + v)      * N + (n0w + 16 + lmod)] = acc01[v] + bv1;
    C[(long)(m0w + 16 + v) * N + (n0w + lmod)]      = acc10[v] + bv0;
    C[(long)(m0w + 16 + v) * N + (n0w + 16 + lmod)] = acc11[v] + bv1;
  }
}

// ---------------------------------------------------------------- RNN scan
// h_t = tanh(XWb[t] + h_{t-1} @ Whh); 1024 threads = 2 per column (split-K).
// Whhp is pack4 f16: Whhp[((i/4)*D + j)*4 + i%4] = Whh[i][j].
__global__ void __launch_bounds__(1024)
k_rnn_scan(const float* __restrict__ XWb, const _Float16* __restrict__ Whhp,
           float* __restrict__ Hout, int nsteps) {
  __shared__ __align__(16) float h[DD];
  __shared__ float part[1024];
  const int tid  = threadIdx.x;
  const int j    = tid & (DD - 1);   // output column
  const int half = tid >> 9;         // 0: i in [0,256), 1: i in [256,512)
  if (half == 0) h[j] = 0.0f;
  __syncthreads();
  const int i0 = half * (DD / 2);
  for (int t = 0; t < nsteps; ++t) {
    if (t + 1 < nsteps)
      __builtin_prefetch(&XWb[(long)(t + 1) * DD + j], 0, 1);  // global_prefetch_b8
    float acc = 0.0f;
#pragma unroll 4
    for (int i = i0; i < i0 + DD / 2; i += 4) {
      float4 hv = *reinterpret_cast<const float4*>(&h[i]);
      H4 w;
      w.u = *reinterpret_cast<const uint2*>(
          &Whhp[((((i >> 2) * DD) + j) << 2)]);  // 8B coalesced: 4 K-values
      acc += hv.x * (float)w.h[0] + hv.y * (float)w.h[1] +
             hv.z * (float)w.h[2] + hv.w * (float)w.h[3];
    }
    part[tid] = acc;
    __syncthreads();                 // all reads of old h done
    if (half == 0) {
      float ht = tanhf(part[j] + part[j + DD] + XWb[(long)t * DD + j]);
      h[j] = ht;
      Hout[(long)t * DD + j] = ht;
    }
    __syncthreads();
  }
}

// ---------------------------------------------------------------- CRF scan
// c'[i] = O[t,i] + Kr[i] + M + log( sum_j ET[i,j] * exp(c[j]-M) ),  M = max(c)
__global__ void __launch_bounds__(1024)
k_crf_scan(const float* __restrict__ O, const _Float16* __restrict__ ETp,
           const float* __restrict__ Kr, float* __restrict__ out) {
  __shared__ __align__(16) float u[VV];
  __shared__ float red[VV];
  __shared__ float part[1024];
  const int tid  = threadIdx.x;
  const int i    = tid & (VV - 1);   // output row
  const int half = tid >> 9;
  float ci = (i == EOS_) ? O[(long)(NSEQ - 1) * VV + EOS_] : -INFINITY;
  const float kri = Kr[i];
  const int j0 = half * (VV / 2);
  for (int t = NSEQ - 2; t >= 0; --t) {
    __builtin_prefetch(&O[(long)t * VV + i], 0, 1);
    if (half == 0) red[i] = ci;
    __syncthreads();
    for (int s = VV / 2; s > 0; s >>= 1) {      // block max over c
      if (tid < s) red[tid] = fmaxf(red[tid], red[tid + s]);
      __syncthreads();
    }
    const float M = red[0];
    if (half == 0) u[i] = __expf(ci - M);       // exp(-inf)=0 handles init
    __syncthreads();
    float acc = 0.0f;
#pragma unroll 4
    for (int jj = j0; jj < j0 + VV / 2; jj += 4) {
      float4 uv = *reinterpret_cast<const float4*>(&u[jj]);
      H4 w;
      w.u = *reinterpret_cast<const uint2*>(
          &ETp[((((jj >> 2) * VV) + i) << 2)]);
      acc += uv.x * (float)w.h[0] + uv.y * (float)w.h[1] +
             uv.z * (float)w.h[2] + uv.w * (float)w.h[3];
    }
    part[tid] = acc;
    __syncthreads();
    if (half == 0)
      ci = O[(long)t * VV + i] + kri + M + __logf(part[i] + part[i + VV]);
    __syncthreads();
  }
  if (tid == BOS_) out[0] = ci;
}

// ---------------------------------------------------------------- launch
extern "C" void kernel_launch(void* const* d_in, const int* in_sizes, int n_in,
                              void* d_out, int out_size, void* d_ws, size_t ws_size,
                              hipStream_t stream) {
  const int*   nums = (const int*)  d_in[0];
  const float* emb  = (const float*)d_in[1];
  const float* Wxh1 = (const float*)d_in[2];
  const float* Whh1 = (const float*)d_in[3];
  const float* b1   = (const float*)d_in[4];
  const float* Wxh2 = (const float*)d_in[5];
  const float* Whh2 = (const float*)d_in[6];
  const float* b2   = (const float*)d_in[7];
  const float* Wl   = (const float*)d_in[8];
  const float* bl   = (const float*)d_in[9];
  const float* T    = (const float*)d_in[10];
  float* out = (float*)d_out;

  // workspace carve-out (~21.6 MB total)
  char* ws = (char*)d_ws;
  size_t off = 0;
  auto carve = [&](size_t bytes) {
    void* p = ws + off;
    off += (bytes + 255) & ~(size_t)255;
    return p;
  };
  float*    XW1   = (float*)   carve((size_t)NSEQ * DD * 4);
  float*    G     = (float*)   carve((size_t)NSEQ * DD * 4);
  float*    XW2   = (float*)   carve((size_t)NSEQ * DD * 4);
  float*    H     = (float*)   carve((size_t)NSEQ * DD * 4);
  float*    O     = (float*)   carve((size_t)NSEQ * VV * 4);
  _Float16* Whh1p = (_Float16*)carve((size_t)DD * DD * 2);
  _Float16* Whh2p = (_Float16*)carve((size_t)DD * DD * 2);
  _Float16* ETp   = (_Float16*)carve((size_t)VV * VV * 2);
  float*    Kr    = (float*)   carve((size_t)VV * 4);
  (void)ws_size; (void)in_sizes; (void)n_in; (void)out_size;

  const int nW = DD * DD;
  k_pack4_f16<<<(nW + 255) / 256, 256, 0, stream>>>(Whh1, Whh1p, DD, DD);
  k_pack4_f16<<<(nW + 255) / 256, 256, 0, stream>>>(Whh2, Whh2p, DD, DD);
  k_crf_prep<<<2, 256, 0, stream>>>(T, ETp, Kr);

  dim3 gGrid(DD / BN, NSEQ / BM);  // (4, 32)
  // layer 1 x-path: emb[nums] @ Wxh1 + b1
  k_gemm_wmma<<<gGrid, 256, 0, stream>>>(emb, nums, Wxh1, b1, XW1, NSEQ, DD, DD);
  k_rnn_scan<<<1, 1024, 0, stream>>>(XW1, Whh1p, G, NSEQ);
  // layer 2 x-path: G @ Wxh2 + b2
  k_gemm_wmma<<<gGrid, 256, 0, stream>>>(G, nullptr, Wxh2, b2, XW2, NSEQ, DD, DD);
  k_rnn_scan<<<1, 1024, 0, stream>>>(XW2, Whh2p, H, NSEQ);
  // emissions: H @ Wl + bl
  k_gemm_wmma<<<gGrid, 256, 0, stream>>>(H, nullptr, Wl, bl, O, NSEQ, VV, DD);
  // CRF backward scan
  k_crf_scan<<<1, 1024, 0, stream>>>(O, ETp, Kr, out);
}
